// LSTMAutoencoder_74328704024750
// MI455X (gfx1250) — matrix operand order repaired and checked by reading, not compiled, not run on hardware
//
#include <hip/hip_runtime.h>

// LSTM autoencoder: B=512, T=1024, encoder 32->64, decoder 64->32, fp32 in/out.
// One WG per 16 batch rows (32 WGs). 6 waves/WG: waves 0-3 encoder (each owns
// 16 hidden units = 4 gate tiles), waves 4-5 decoder (each owns 16 of 32 units).
// Weights as bf16 WMMA B-fragments resident in VGPRs; WMMA chains start from
// inline-constant C=0 (no accumulator init movs); bias folded into the
// activation math; c-state in accumulator registers; h exchanged via
// double-buffered LDS A-matrices; decoder runs one step behind encoder
// (software pipeline), one barrier per timestep; gfx1250 v_tanh_f32.

#define B_SZ 512
#define T_SZ 1024
#define F_IN 32
#define H_ENC 64

typedef __attribute__((ext_vector_type(16))) __bf16 v16bf;
typedef __attribute__((ext_vector_type(8)))  float v8f;
typedef __attribute__((ext_vector_type(4)))  float v4f;   // native vector for NT loads

union Frag32B { uint4 q[2]; v16bf v; };

#if __has_builtin(__builtin_amdgcn_tanhf)
__device__ __forceinline__ float tanh_fast(float x) {
  return __builtin_amdgcn_tanhf(x);           // v_tanh_f32 (gfx1250 TRANS op)
}
#else
__device__ __forceinline__ float tanh_fast(float x) {
  const float e = __builtin_amdgcn_exp2f(-2.88539008177793f * x); // exp(-2x)
  return 2.0f * __builtin_amdgcn_rcpf(1.0f + e) - 1.0f;
}
#endif
// sigmoid(x + b) with hb = 0.5*b precomputed: fma + tanh + fma.
__device__ __forceinline__ float sigmoid_b(float x, float hb) {
  return __builtin_fmaf(0.5f, tanh_fast(__builtin_fmaf(0.5f, x, hb)), 0.5f);
}

__global__ __launch_bounds__(192, 1)
void lstm_autoenc_cdna5(const float* __restrict__ x,
                        const float* __restrict__ ew_ih, const float* __restrict__ ew_hh,
                        const float* __restrict__ eb_ih, const float* __restrict__ eb_hh,
                        const float* __restrict__ dw_ih, const float* __restrict__ dw_hh,
                        const float* __restrict__ db_ih, const float* __restrict__ db_hh,
                        float* __restrict__ out)
{
  // Double-buffered A matrices (row-major [16 rows][96 K] bf16).
  // Aenc: [ x_t (K 0..31) | h_enc_{t-1} (K 32..95) ]
  // Adec: [ h_enc_t (K 0..63) | h_dec_{t-1} (K 64..95) ]
  __shared__ __align__(16) __bf16 Aenc[2][16][96];
  __shared__ __align__(16) __bf16 Adec[2][16][96];

  const int tid  = threadIdx.x;
  const int wave = tid >> 5;
  const int lane = tid & 31;
  const int ln16 = lane & 15;
  const int hiL  = lane >> 4;            // 0|1: selects K-halves in WMMA layouts
  const int b0   = blockIdx.x * 16;      // batch rows owned by this WG

  const bool isEnc = (wave < 4);
  const int  uw    = isEnc ? wave : (wave - 4);   // 16-unit block index

  // ------- loop-invariant B fragments (bf16) + per-lane bias scalars -------
  // B layout used here (contiguous k, same as A): lane<16 holds K=0..15 (elem e -> K=e),
  // lane>=16 holds K=16..31.  N = lane&15.
  v16bf Bf[4][3];
  float hb[4];     // gates i,f,o: 0.5*bias; gate g (idx 2): full bias
  if (isEnc) {
    const int n = uw * 16 + ln16;                 // unit within 64
#pragma unroll
    for (int g = 0; g < 4; ++g) {
      const int col = g * 64 + n;                 // gate column 0..255 (i,f,g,o)
      const float bsum = eb_ih[col] + eb_hh[col];
      hb[g] = (g == 2) ? bsum : 0.5f * bsum;
#pragma unroll
      for (int kc = 0; kc < 3; ++kc) {
        v16bf bf;
#pragma unroll
        for (int e = 0; e < 16; ++e) {
          const int K = kc * 32 + hiL * 16 + e;   // 0..95
          const float v = (K < 32) ? ew_ih[col * F_IN + K]
                                   : ew_hh[col * H_ENC + (K - 32)];
          bf[e] = (__bf16)v;
        }
        Bf[g][kc] = bf;
      }
    }
  } else {
    const int n = uw * 16 + ln16;                 // unit within 32
#pragma unroll
    for (int g = 0; g < 4; ++g) {
      const int col = g * 32 + n;                 // gate column 0..127
      const float bsum = db_ih[col] + db_hh[col];
      hb[g] = (g == 2) ? bsum : 0.5f * bsum;
#pragma unroll
      for (int kc = 0; kc < 3; ++kc) {
        v16bf bf;
#pragma unroll
        for (int e = 0; e < 16; ++e) {
          const int K = kc * 32 + hiL * 16 + e;   // 0..95
          const float v = (K < 64) ? dw_ih[col * H_ENC + K]
                                   : dw_hh[col * F_IN + (K - 64)];
          bf[e] = (__bf16)v;
        }
        Bf[g][kc] = bf;
      }
    }
  }

  // ------- prologue: zero LDS (h_{-1}=0), stage x_0 -------
  {
    __bf16* p0 = &Aenc[0][0][0];
    __bf16* p1 = &Adec[0][0][0];
    for (int i = tid; i < 2 * 16 * 96; i += 192) { p0[i] = (__bf16)0.0f; p1[i] = (__bf16)0.0f; }
  }
  __syncthreads();
  if (isEnc) {
    const int xrow = uw * 4 + (lane >> 3);        // 4 waves x 4 rows = 16 rows
    const int f0   = (lane & 7) * 4;              // 8 lanes x 4 feats = 32
    const v4f xv =
        __builtin_nontemporal_load((const v4f*)&x[((size_t)(b0 + xrow) * T_SZ) * F_IN + f0]);
    __bf16* d = &Aenc[0][xrow][f0];
    d[0] = (__bf16)xv[0]; d[1] = (__bf16)xv[1]; d[2] = (__bf16)xv[2]; d[3] = (__bf16)xv[3];
  }
  __syncthreads();

  v8f cstate = {};   // cell state, same 16x16 layout as the WMMA accumulator

  // ------- recurrence: encoder does t, decoder does t-1 (pipelined) -------
  for (int t = 0; t <= T_SZ; ++t) {
    const int pt = t & 1;
    const int nt = pt ^ 1;

    if (isEnc) {
      if (t < T_SZ) {
        // Load x_{t+1} early (latency hidden behind this step's math).
        const int  xrow    = uw * 4 + (lane >> 3);
        const int  f0      = (lane & 7) * 4;
        const bool hasNext = (t + 1) < T_SZ;
        v4f xv = {0.f, 0.f, 0.f, 0.f};
        if (hasNext)
          xv = __builtin_nontemporal_load(
              (const v4f*)&x[((size_t)(b0 + xrow) * T_SZ + (t + 1)) * F_IN + f0]);
        if ((t + 2) < T_SZ)
          __builtin_prefetch(&x[((size_t)(b0 + xrow) * T_SZ + (t + 2)) * F_IN + f0], 0, 1);

        // A fragments (16x32 bf16 chunks) from LDS: two b128 loads each.
        const int arow = ln16;
        const int kb   = hiL * 16;
        Frag32B f0a, f1a, f2a;
        f0a.q[0] = *(const uint4*)&Aenc[pt][arow][0 * 32 + kb];
        f0a.q[1] = *(const uint4*)&Aenc[pt][arow][0 * 32 + kb + 8];
        f1a.q[0] = *(const uint4*)&Aenc[pt][arow][1 * 32 + kb];
        f1a.q[1] = *(const uint4*)&Aenc[pt][arow][1 * 32 + kb + 8];
        f2a.q[0] = *(const uint4*)&Aenc[pt][arow][2 * 32 + kb];
        f2a.q[1] = *(const uint4*)&Aenc[pt][arow][2 * 32 + kb + 8];
        const v16bf a0 = f0a.v, a1 = f1a.v, a2 = f2a.v;

        // 4 gate tiles (i,f,g,o share the (row,unit) C-layout), K=96 = 3 WMMAs.
        // First WMMA of each chain uses C = literal zero -> inline SRC2=0.
        v8f acc[4];
#pragma unroll
        for (int g = 0; g < 4; ++g) {
          v8f c = {};
          c = __builtin_amdgcn_wmma_f32_16x16x32_bf16(false, a0, false, Bf[g][0], (short)0, c, false, false);
          c = __builtin_amdgcn_wmma_f32_16x16x32_bf16(false, a1, false, Bf[g][1], (short)0, c, false, false);
          c = __builtin_amdgcn_wmma_f32_16x16x32_bf16(false, a2, false, Bf[g][2], (short)0, c, false, false);
          acc[g] = c;
        }

        // Gate activations (bias folded in) + state update; publish h_enc[t].
        const int unit = uw * 16 + ln16;
#pragma unroll
        for (int r = 0; r < 8; ++r) {
          const float ig = sigmoid_b(acc[0][r], hb[0]);
          const float fg = sigmoid_b(acc[1][r], hb[1]);
          const float gg = tanh_fast(acc[2][r] + hb[2]);
          const float og = sigmoid_b(acc[3][r], hb[3]);
          const float cc = __builtin_fmaf(fg, cstate[r], ig * gg);
          cstate[r] = cc;
          const float h  = og * tanh_fast(cc);
          const int orow = r + 8 * hiL;
          const __bf16 hbv = (__bf16)h;
          Aenc[nt][orow][32 + unit] = hbv;  // next encoder A
          Adec[pt][orow][unit]      = hbv;  // decoder A for step t
        }
        if (hasNext) {
          __bf16* d = &Aenc[nt][xrow][f0];
          d[0] = (__bf16)xv[0]; d[1] = (__bf16)xv[1]; d[2] = (__bf16)xv[2]; d[3] = (__bf16)xv[3];
        }
      }
    } else {
      if (t >= 1) {                       // decoder step td = t-1 reads Adec[nt]
        const int td   = t - 1;
        const int arow = ln16;
        const int kb   = hiL * 16;
        Frag32B f0a, f1a, f2a;
        f0a.q[0] = *(const uint4*)&Adec[nt][arow][0 * 32 + kb];
        f0a.q[1] = *(const uint4*)&Adec[nt][arow][0 * 32 + kb + 8];
        f1a.q[0] = *(const uint4*)&Adec[nt][arow][1 * 32 + kb];
        f1a.q[1] = *(const uint4*)&Adec[nt][arow][1 * 32 + kb + 8];
        f2a.q[0] = *(const uint4*)&Adec[nt][arow][2 * 32 + kb];
        f2a.q[1] = *(const uint4*)&Adec[nt][arow][2 * 32 + kb + 8];
        const v16bf a0 = f0a.v, a1 = f1a.v, a2 = f2a.v;

        v8f acc[4];
#pragma unroll
        for (int g = 0; g < 4; ++g) {
          v8f c = {};
          c = __builtin_amdgcn_wmma_f32_16x16x32_bf16(false, a0, false, Bf[g][0], (short)0, c, false, false);
          c = __builtin_amdgcn_wmma_f32_16x16x32_bf16(false, a1, false, Bf[g][1], (short)0, c, false, false);
          c = __builtin_amdgcn_wmma_f32_16x16x32_bf16(false, a2, false, Bf[g][2], (short)0, c, false, false);
          acc[g] = c;
        }

        const int unit = uw * 16 + ln16;   // decoder hidden unit == output feature
#pragma unroll
        for (int r = 0; r < 8; ++r) {
          const float ig = sigmoid_b(acc[0][r], hb[0]);
          const float fg = sigmoid_b(acc[1][r], hb[1]);
          const float gg = tanh_fast(acc[2][r] + hb[2]);
          const float og = sigmoid_b(acc[3][r], hb[3]);
          const float cc = __builtin_fmaf(fg, cstate[r], ig * gg);
          cstate[r] = cc;
          const float h  = og * tanh_fast(cc);
          const int orow = r + 8 * hiL;
          Adec[pt][orow][64 + unit] = (__bf16)h;     // h_dec for step td+1
          __builtin_nontemporal_store(
              h, &out[((size_t)(b0 + orow) * T_SZ + td) * F_IN + unit]);
        }
      }
    }
    __syncthreads();
  }
}

extern "C" void kernel_launch(void* const* d_in, const int* in_sizes, int n_in,
                              void* d_out, int out_size, void* d_ws, size_t ws_size,
                              hipStream_t stream) {
  const float* x     = (const float*)d_in[0];
  const float* ew_ih = (const float*)d_in[1];
  const float* ew_hh = (const float*)d_in[2];
  const float* eb_ih = (const float*)d_in[3];
  const float* eb_hh = (const float*)d_in[4];
  const float* dw_ih = (const float*)d_in[5];
  const float* dw_hh = (const float*)d_in[6];
  const float* db_ih = (const float*)d_in[7];
  const float* db_hh = (const float*)d_in[8];
  float* out = (float*)d_out;

  dim3 grid(B_SZ / 16);   // 32 independent recurrence chains
  dim3 block(192);        // 6 waves: 4 encoder + 2 decoder
  lstm_autoenc_cdna5<<<grid, block, 0, stream>>>(
      x, ew_ih, ew_hh, eb_ih, eb_hh, dw_ih, dw_hh, db_ih, db_hh, out);
}
